// HSSFN_11416023072918
// MI455X (gfx1250) — compile-verified, matrix operand order checked by reading, not run on hardware
//
#include <hip/hip_runtime.h>
#include <hip/hip_bf16.h>
#include <math.h>

typedef __attribute__((ext_vector_type(16))) __bf16 v16bf;
typedef __attribute__((ext_vector_type(8)))  float  v8f;

#define BN_TOK 8192   // B*N
#define NTOK_PER_B 1024
#define DMODEL 256
#define DINNER 512
#define DSTATE 16
#define DXDBL 48      // DTR + 2*DS

__device__ __forceinline__ __bf16 f2bf(float x) {
  union { __hip_bfloat16 h; __bf16 b; } u;
  u.h = __float2bfloat16(x);
  return u.b;
}

// pack 8 consecutive floats -> 8 bf16 in a uint4 (2 bf16 per dword)
__device__ __forceinline__ uint4 pack8(const float* v) {
  union { __bf16 b[8]; uint4 q; } u;
  #pragma unroll
  for (int j = 0; j < 8; ++j) u.b[j] = f2bf(v[j]);
  return u.q;
}

// load 16 consecutive floats (vectorized) or zero-fill
__device__ __forceinline__ void load16(const float* src, bool ok, float* f) {
  if (ok) {
    *(float4*)(f)      = *(const float4*)(src);
    *(float4*)(f + 4)  = *(const float4*)(src + 4);
    *(float4*)(f + 8)  = *(const float4*)(src + 8);
    *(float4*)(f + 12) = *(const float4*)(src + 12);
  } else {
    #pragma unroll
    for (int j = 0; j < 16; ++j) f[j] = 0.0f;
  }
}

__device__ __forceinline__ void store16(__bf16* dst, const float* f) {
  *(uint4*)(dst)     = pack8(f);
  *(uint4*)(dst + 8) = pack8(f + 8);
}

template <int ACT>
__device__ __forceinline__ float activate(float x) {
  if constexpr (ACT == 1) {               // exact GELU
    return 0.5f * x * (1.0f + erff(x * 0.70710678118654752f));
  } else if constexpr (ACT == 2) {        // sigmoid
    return 1.0f / (1.0f + __expf(-x));
  } else {
    return x;
  }
}

// ---------------------------------------------------------------------------
// Universal WMMA GEMM: C[M,N] = act(A[M,K] @ W[N,K]^T + bias) * scale + addend
// REQUIRES K % 32 == 0. 256 threads = 8 waves; block tile 128(M) x 128(N).
// LDS double-buffered; next k-chunk prefetched into registers during WMMA.
// ---------------------------------------------------------------------------
template <int ACT>
__global__ __launch_bounds__(256) void gemm_wmma(
    const float* __restrict__ A, int lda,
    const float* __restrict__ W,          // [N,K] row-major
    const float* __restrict__ bias,       // [N] or null
    const float* __restrict__ scale,      // [N] or null
    const float* __restrict__ addend,     // [M,*] or null
    int ld_add,
    float* __restrict__ C, int ldc,
    int M, int N, int K) {
  __shared__ __align__(16) __bf16 As[2][128][40];
  __shared__ __align__(16) __bf16 Ws[2][128][40];

  const int tid  = threadIdx.x;
  const int lane = tid & 31;
  const int wave = tid >> 5;
  const int wm   = wave >> 1;     // 0..3  (M sub-block of 32)
  const int wn   = wave & 1;      // 0..1  (N sub-block of 64)
  const int m_block = blockIdx.y * 128;
  const int n_block = blockIdx.x * 128;

  v8f acc[2][4] = {};

  const int r  = lane & 15;
  const int kb = (lane >> 4) * 8;

  const int  srow = tid >> 1;
  const int  scb  = (tid & 1) * 16;
  const bool aOk  = (m_block + srow) < M;
  const bool wOk  = (n_block + srow) < N;
  const float* aSrc = A + (long)(m_block + srow) * lda + scb;
  const float* wSrc = W + (long)(n_block + srow) * K + scb;

  // prologue: stage k-chunk 0 into buffer 0
  {
    float fA[16], fW[16];
    load16(aSrc, aOk, fA);
    load16(wSrc, wOk, fW);
    store16(&As[0][srow][scb], fA);
    store16(&Ws[0][srow][scb], fW);
  }
  __syncthreads();

  int buf = 0;
  for (int kk = 0; kk < K; kk += 32) {
    const bool hasNext = (kk + 32 < K);
    float nA[16], nW[16];
    if (hasNext) {                        // issue next chunk's global loads
      load16(aSrc + kk + 32, aOk, nA);
      load16(wSrc + kk + 32, wOk, nW);
    }

    // ---- fragments per ISA 16-bit 16x32 A layout -------------------------
    const __bf16 (*Ab)[40] = As[buf];
    const __bf16 (*Wb)[40] = Ws[buf];
    v16bf afrag[2], bfrag[4];
    #pragma unroll
    for (int i = 0; i < 2; ++i) {
      int arow = wm * 32 + i * 16 + r;
      #pragma unroll
      for (int j = 0; j < 8; ++j) {
        afrag[i][j]     = Ab[arow][kb + j];
        afrag[i][j + 8] = Ab[arow][kb + 16 + j];
      }
    }
    #pragma unroll
    for (int i = 0; i < 4; ++i) {
      int brow = wn * 64 + i * 16 + r;
      #pragma unroll
      for (int j = 0; j < 8; ++j) {
        bfrag[i][j]     = Wb[brow][kb + j];
        bfrag[i][j + 8] = Wb[brow][kb + 16 + j];
      }
    }

    #pragma unroll
    for (int i = 0; i < 2; ++i)
      #pragma unroll
      for (int j = 0; j < 4; ++j)
        acc[i][j] = __builtin_amdgcn_wmma_f32_16x16x32_bf16(
            false, afrag[i], false, bfrag[j], (short)0, acc[i][j], false, false);

    if (hasNext) {                        // store prefetched chunk, flip
      store16(&As[buf ^ 1][srow][scb], nA);
      store16(&Ws[buf ^ 1][srow][scb], nW);
      buf ^= 1;
      __syncthreads();
    }
  }

  // ---- epilogue: VGPR rr holds M=rr (lanes<16) / M=8+rr (lanes>=16) -------
  const int mhalf = (lane >= 16) ? 8 : 0;
  const int ncl   = lane & 15;
  #pragma unroll
  for (int i = 0; i < 2; ++i) {
    #pragma unroll
    for (int j = 0; j < 4; ++j) {
      int ncol = n_block + wn * 64 + j * 16 + ncl;
      if (ncol >= N) continue;
      float bv = bias  ? bias[ncol]  : 0.0f;
      float sv = scale ? scale[ncol] : 1.0f;
      #pragma unroll
      for (int rr = 0; rr < 8; ++rr) {
        int m = m_block + wm * 32 + i * 16 + mhalf + rr;
        if (m < M) {
          float v = activate<ACT>(acc[i][j][rr] + bv) * sv;
          if (addend) v += addend[(long)m * ld_add + ncol];
          C[(long)m * ldc + ncol] = v;
        }
      }
    }
  }
}

// ---------------------------------------------------------------------------
// delta = softplus(xdbl[:, :16] @ dt_w^T + dt_b)   (K=16: plain VALU kernel)
// one block per token; dt row cached in LDS.
// ---------------------------------------------------------------------------
__global__ __launch_bounds__(256) void dt_kernel(
    const float* __restrict__ xdbl, const float* __restrict__ dt_w,
    const float* __restrict__ dt_b, float* __restrict__ delta) {
  __shared__ float dtr[16];
  long tok = blockIdx.x;
  if (threadIdx.x < 16) dtr[threadIdx.x] = xdbl[tok * DXDBL + threadIdx.x];
  __syncthreads();
  #pragma unroll
  for (int rep = 0; rep < 2; ++rep) {
    int col = threadIdx.x + rep * 256;
    float acc = dt_b[col];
    #pragma unroll
    for (int k = 0; k < 16; ++k) acc += dtr[k] * dt_w[col * 16 + k];
    delta[tok * DINNER + col] = (acc > 20.0f) ? acc : log1pf(__expf(acc));
  }
}

// ---------------------------------------------------------------------------
// LayerNorm over D=256, one wave per token (8 elems/lane).
// ---------------------------------------------------------------------------
__global__ __launch_bounds__(256) void ln_kernel(
    const float* __restrict__ in, const float* __restrict__ g,
    const float* __restrict__ b, float* __restrict__ out, int ntok) {
  int gt   = blockIdx.x * blockDim.x + threadIdx.x;
  int tok  = gt >> 5;
  int lane = gt & 31;
  if (tok >= ntok) return;
  const float* row = in + (long)tok * DMODEL;
  float v[8];
  float s = 0.0f;
  #pragma unroll
  for (int j = 0; j < 8; ++j) { v[j] = row[lane + 32 * j]; s += v[j]; }
  #pragma unroll
  for (int m = 16; m >= 1; m >>= 1) s += __shfl_xor(s, m, 32);
  float mean = s * (1.0f / 256.0f);
  float var = 0.0f;
  #pragma unroll
  for (int j = 0; j < 8; ++j) { float d = v[j] - mean; var += d * d; }
  #pragma unroll
  for (int m = 16; m >= 1; m >>= 1) var += __shfl_xor(var, m, 32);
  var *= (1.0f / 256.0f);
  float inv = rsqrtf(var + 1e-5f);
  #pragma unroll
  for (int j = 0; j < 8; ++j) {
    int c = lane + 32 * j;
    out[(long)tok * DMODEL + c] = (v[j] - mean) * inv * g[c] + b[c];
  }
}

// ---------------------------------------------------------------------------
// Causal depthwise conv (width 4) + SiLU.  xin = xz[:, 0:512].
// ---------------------------------------------------------------------------
__global__ __launch_bounds__(256) void conv_silu(
    const float* __restrict__ xz, const float* __restrict__ cw,
    const float* __restrict__ cb, float* __restrict__ xc, long total) {
  long i = (long)blockIdx.x * blockDim.x + threadIdx.x;
  if (i >= total) return;
  int  d   = (int)(i & 511);
  long tok = i >> 9;
  int  n   = (int)(tok & 1023);
  long bb  = tok >> 10;
  float acc = cb[d];
  #pragma unroll
  for (int k = 0; k < 4; ++k) {
    int nn = n - 3 + k;
    if (nn >= 0) acc += cw[d * 4 + k] * xz[((bb << 10) + nn) * 1024 + d];
  }
  xc[tok * DINNER + d] = acc / (1.0f + __expf(-acc));
}

// ---------------------------------------------------------------------------
// Selective scan: one (b,d) chain per 16 lanes, s = lane&15, 1024 steps.
// ---------------------------------------------------------------------------
__global__ __launch_bounds__(256) void mamba_scan(
    const float* __restrict__ delta, const float* __restrict__ xc,
    const float* __restrict__ xdbl, const float* __restrict__ A_log,
    const float* __restrict__ Dp, float* __restrict__ y) {
  int tid   = blockIdx.x * blockDim.x + threadIdx.x;
  int chain = tid >> 4;
  int s     = tid & 15;
  if (chain >= 8 * DINNER) return;
  int d  = chain & (DINNER - 1);
  int bb = chain >> 9;
  float a      = -__expf(A_log[d * DSTATE + s]);
  float dscale = Dp[d];
  float h = 0.0f;
  long tokbase = (long)bb * NTOK_PER_B;
  for (int n = 0; n < NTOK_PER_B; ++n) {
    long tok  = tokbase + n;
    float dlt = delta[tok * DINNER + d];
    float xv  = xc[tok * DINNER + d];
    float bm  = xdbl[tok * DXDBL + 16 + s];
    float cm  = xdbl[tok * DXDBL + 32 + s];
    h = __expf(dlt * a) * h + dlt * bm * xv;
    float p = h * cm;
    p += __shfl_xor(p, 1, 32);
    p += __shfl_xor(p, 2, 32);
    p += __shfl_xor(p, 4, 32);
    p += __shfl_xor(p, 8, 32);
    if (s == 0) y[tok * DINNER + d] = p + xv * dscale;
  }
}

// y *= silu(z),  z = xz[:, 512:1024]
__global__ __launch_bounds__(256) void gate_silu(
    float* __restrict__ y, const float* __restrict__ xz, long total) {
  long i = (long)blockIdx.x * blockDim.x + threadIdx.x;
  if (i >= total) return;
  int  d   = (int)(i & 511);
  long tok = i >> 9;
  float z = xz[tok * 1024 + 512 + d];
  y[i] *= z / (1.0f + __expf(-z));
}

// mean over tokens per batch: ms[B,N,768] -> mean[B,768]; grid = 24 blocks
__global__ __launch_bounds__(256) void mean_tokens(
    const float* __restrict__ ms, float* __restrict__ mean) {
  int b  = blockIdx.x / 3;
  int c  = (blockIdx.x % 3) * 256 + threadIdx.x;
  float s = 0.0f;
  for (int n = 0; n < NTOK_PER_B; ++n)
    s += ms[((long)b * NTOK_PER_B + n) * 768 + c];
  mean[b * 768 + c] = s * (1.0f / 1024.0f);
}

// aggregator: h = gelu(mean @ w1^T + b1); c = softmax(h@w2^T+b2); c = softmax(c/(t+1e-6))
__global__ __launch_bounds__(256) void agg_kernel(
    const float* __restrict__ mean, const float* __restrict__ w1,
    const float* __restrict__ b1, const float* __restrict__ w2,
    const float* __restrict__ b2, const float* __restrict__ temp,
    float* __restrict__ cout) {
  __shared__ float h[256];
  __shared__ float cl[3];
  int t = threadIdx.x;
  for (int b = 0; b < 8; ++b) {
    float acc = b1[t];
    for (int k = 0; k < 768; ++k) acc += mean[b * 768 + k] * w1[t * 768 + k];
    h[t] = 0.5f * acc * (1.0f + erff(acc * 0.70710678118654752f));
    __syncthreads();
    if (t < 3) {
      float a2 = b2[t];
      for (int k = 0; k < 256; ++k) a2 += h[k] * w2[t * 256 + k];
      cl[t] = a2;
    }
    __syncthreads();
    if (t == 0) {
      float m = fmaxf(cl[0], fmaxf(cl[1], cl[2]));
      float e0 = __expf(cl[0] - m), e1 = __expf(cl[1] - m), e2 = __expf(cl[2] - m);
      float S = e0 + e1 + e2;
      float it = 1.0f / (temp[0] + 1e-6f);
      float q0 = (e0 / S) * it, q1 = (e1 / S) * it, q2 = (e2 / S) * it;
      float m2 = fmaxf(q0, fmaxf(q1, q2));
      float f0 = __expf(q0 - m2), f1 = __expf(q1 - m2), f2 = __expf(q2 - m2);
      float S2 = f0 + f1 + f2;
      cout[b * 3 + 0] = f0 / S2;
      cout[b * 3 + 1] = f1 / S2;
      cout[b * 3 + 2] = f2 / S2;
    }
    __syncthreads();
  }
}

// w = c0*ep + c1*bp + c2*sp  (ep/bp/sp are ms column blocks)
__global__ __launch_bounds__(256) void mix_kernel(
    const float* __restrict__ ms, const float* __restrict__ c,
    float* __restrict__ w, long total) {
  long i = (long)blockIdx.x * blockDim.x + threadIdx.x;
  if (i >= total) return;
  int  col = (int)(i & 255);
  long tok = i >> 8;
  int  b   = (int)(tok >> 10);
  const float* cr  = c + b * 3;
  const float* row = ms + tok * 768;
  w[i] = cr[0] * row[col] + cr[1] * row[256 + col] + cr[2] * row[512 + col];
}

// out = a + s[col]*b
__global__ __launch_bounds__(256) void axpy_kernel(
    const float* __restrict__ a, const float* __restrict__ s,
    const float* __restrict__ bsrc, float* __restrict__ out, long total) {
  long i = (long)blockIdx.x * blockDim.x + threadIdx.x;
  if (i >= total) return;
  out[i] = a[i] + s[i & 255] * bsrc[i];
}

// row softmax over 64 cols with pre-scale 1/16; one wave per row, in-place
__global__ __launch_bounds__(256) void softmax64(float* __restrict__ scores, int rows) {
  int gt   = blockIdx.x * blockDim.x + threadIdx.x;
  int row  = gt >> 5;
  int lane = gt & 31;
  if (row >= rows) return;
  float* r = scores + (long)row * 64;
  float a = r[lane] * 0.0625f;
  float b = r[lane + 32] * 0.0625f;
  float m = fmaxf(a, b);
  #pragma unroll
  for (int mm = 16; mm >= 1; mm >>= 1) m = fmaxf(m, __shfl_xor(m, mm, 32));
  float ea = __expf(a - m), eb = __expf(b - m);
  float s = ea + eb;
  #pragma unroll
  for (int mm = 16; mm >= 1; mm >>= 1) s += __shfl_xor(s, mm, 32);
  float inv = 1.0f / s;
  r[lane]      = ea * inv;
  r[lane + 32] = eb * inv;
}

// transpose proj [64,256] -> projT [256,64]
__global__ __launch_bounds__(256) void transpose64x256(
    const float* __restrict__ src, float* __restrict__ dst) {
  int i = blockIdx.x * blockDim.x + threadIdx.x;
  if (i >= 64 * 256) return;
  int r = i >> 8, c = i & 255;
  dst[c * 64 + r] = src[i];
}

// x2 = x + ls2*( (xn + gate*ctx) - x )
__global__ __launch_bounds__(256) void pcm_combine(
    const float* __restrict__ x, const float* __restrict__ xn,
    const float* __restrict__ gate, const float* __restrict__ ctx,
    const float* __restrict__ ls2, float* __restrict__ out, long total) {
  long i = (long)blockIdx.x * blockDim.x + threadIdx.x;
  if (i >= total) return;
  int c = (int)(i & 255);
  float coh = xn[i] + gate[i] * ctx[i];
  out[i] = x[i] + ls2[c] * (coh - x[i]);
}

// ---------------------------------------------------------------------------
// Host-side GEMM dispatch (K must be a multiple of 32)
// ---------------------------------------------------------------------------
static void launch_gemm(int act, const float* A, int lda, const float* W,
                        const float* bias, const float* scale,
                        const float* addend, int ld_add, float* C, int ldc,
                        int M, int N, int K, hipStream_t stream) {
  dim3 grid((N + 127) / 128, (M + 127) / 128);
  dim3 block(256);
  switch (act) {
    case 1:
      gemm_wmma<1><<<grid, block, 0, stream>>>(A, lda, W, bias, scale, addend, ld_add, C, ldc, M, N, K);
      break;
    case 2:
      gemm_wmma<2><<<grid, block, 0, stream>>>(A, lda, W, bias, scale, addend, ld_add, C, ldc, M, N, K);
      break;
    default:
      gemm_wmma<0><<<grid, block, 0, stream>>>(A, lda, W, bias, scale, addend, ld_add, C, ldc, M, N, K);
      break;
  }
}

extern "C" void kernel_launch(void* const* d_in, const int* in_sizes, int n_in,
                              void* d_out, int out_size, void* d_ws, size_t ws_size,
                              hipStream_t stream) {
  const int BN = BN_TOK;
  const long EL = (long)BN * DMODEL;
  const int EB = (int)((EL + 255) / 256);

  // ---- input pointers (setup_inputs dict order, leaves flattened) ---------
  const float* edge = (const float*)d_in[0];
  const float* blob = (const float*)d_in[1];
  const float* spec = (const float*)d_in[2];
  const float* ln1_g = (const float*)d_in[3];
  const float* ln1_b = (const float*)d_in[4];
  const float* ln2_g = (const float*)d_in[5];
  const float* ln2_b = (const float*)d_in[6];
  const float* ln3_g = (const float*)d_in[7];
  const float* ln3_b = (const float*)d_in[8];
  const float* lnX_g[3] = {(const float*)d_in[9], (const float*)d_in[11], (const float*)d_in[13]};
  const float* lnX_b[3] = {(const float*)d_in[10], (const float*)d_in[12], (const float*)d_in[14]};
  const int mbase[3] = {15, 24, 33};
  const float* temp   = (const float*)d_in[42];
  const float* agg_w1 = (const float*)d_in[43];
  const float* agg_b1 = (const float*)d_in[44];
  const float* agg_w2 = (const float*)d_in[45];
  const float* agg_b2 = (const float*)d_in[46];
  const float* ssf_out_w = (const float*)d_in[47];
  const float* ssf_out_b = (const float*)d_in[48];
  const float* protos  = (const float*)d_in[49];
  const float* proj_w  = (const float*)d_in[50];
  const float* proj_b  = (const float*)d_in[51];
  const float* gate_w  = (const float*)d_in[52];
  const float* gate_b  = (const float*)d_in[53];
  const float* ffn_w1  = (const float*)d_in[54];
  const float* ffn_b1  = (const float*)d_in[55];
  const float* ffn_w2  = (const float*)d_in[56];
  const float* ffn_b2  = (const float*)d_in[57];
  const float* ls1 = (const float*)d_in[58];
  const float* ls2 = (const float*)d_in[59];
  const float* ls3 = (const float*)d_in[60];

  // ---- workspace carve-out (floats) --------------------------------------
  float* p    = (float*)d_ws;
  float* ms   = p;  p += (long)768 * BN;
  float* X0   = p;  p += (long)256 * BN;
  float* Xn   = p;  p += (long)256 * BN;   // mamba-input LN; later ln2-out / ln3-out
  float* wbuf = p;  p += (long)256 * BN;   // mix buffer; later gate
  float* fus  = p;  p += (long)256 * BN;   // fusion; later ctx
  float* xbuf = p;  p += (long)256 * BN;
  float* x2   = p;  p += (long)256 * BN;
  float* big  = p;
  float* xz    = big;                      // 1024*BN
  float* xc    = big + (long)1024 * BN;    //  512*BN
  float* xdbl  = xc  + (long)512 * BN;     //   48*BN
  float* delta = xdbl + (long)48 * BN;     //  512*BN
  float* ybuf  = delta + (long)512 * BN;   //  512*BN
  float* hbuf   = big;                     // reuse (ffn hidden, 1024*BN)
  float* scores = big + (long)1024 * BN;   // reuse (64*BN)
  p = ybuf + (long)512 * BN;
  float* projb = p;  p += 64 * 256;
  float* projT = p;  p += 64 * 256;
  float* meanb = p;  p += 8 * 768;
  float* cbuf  = p;  p += 32;

  const float* inputs[3] = {edge, blob, spec};

  // ================= three Mamba branches =================================
  for (int s = 0; s < 3; ++s) {
    const float* in_w    = (const float*)d_in[mbase[s] + 0];
    const float* conv_w  = (const float*)d_in[mbase[s] + 1];
    const float* conv_b  = (const float*)d_in[mbase[s] + 2];
    const float* xproj_w = (const float*)d_in[mbase[s] + 3];
    const float* dt_w    = (const float*)d_in[mbase[s] + 4];
    const float* dt_b    = (const float*)d_in[mbase[s] + 5];
    const float* A_log   = (const float*)d_in[mbase[s] + 6];
    const float* Dp      = (const float*)d_in[mbase[s] + 7];
    const float* out_w   = (const float*)d_in[mbase[s] + 8];

    ln_kernel<<<BN / 8, 256, 0, stream>>>(inputs[s], ln1_g, ln1_b, X0, BN);
    ln_kernel<<<BN / 8, 256, 0, stream>>>(X0, lnX_g[s], lnX_b[s], Xn, BN);
    launch_gemm(0, Xn, 256, in_w, nullptr, nullptr, nullptr, 0, xz, 1024, BN, 1024, 256, stream);
    conv_silu<<<(int)(((long)BN * 512 + 255) / 256), 256, 0, stream>>>(xz, conv_w, conv_b, xc, (long)BN * 512);
    launch_gemm(0, xc, 512, xproj_w, nullptr, nullptr, nullptr, 0, xdbl, 48, BN, 48, 512, stream);
    dt_kernel<<<BN, 256, 0, stream>>>(xdbl, dt_w, dt_b, delta);
    mamba_scan<<<256, 256, 0, stream>>>(delta, xc, xdbl, A_log, Dp, ybuf);
    gate_silu<<<(int)(((long)BN * 512 + 255) / 256), 256, 0, stream>>>(ybuf, xz, (long)BN * 512);
    launch_gemm(0, ybuf, 512, out_w, nullptr, nullptr, X0, 256, ms + s * 256, 768, BN, 256, 512, stream);
  }

  // ================= fusion ===============================================
  mean_tokens<<<24, 256, 0, stream>>>(ms, meanb);
  agg_kernel<<<1, 256, 0, stream>>>(meanb, agg_w1, agg_b1, agg_w2, agg_b2, temp, cbuf);
  mix_kernel<<<EB, 256, 0, stream>>>(ms, cbuf, wbuf, EL);
  launch_gemm(0, ms, 768, ssf_out_w, ssf_out_b, nullptr, wbuf, 256, fus, 256, BN, 256, 768, stream);
  axpy_kernel<<<EB, 256, 0, stream>>>(spec, ls1, fus, xbuf, EL);

  // ================= prototype coherence ==================================
  ln_kernel<<<BN / 8, 256, 0, stream>>>(xbuf, ln2_g, ln2_b, Xn, BN);
  launch_gemm(0, protos, 256, proj_w, proj_b, nullptr, nullptr, 0, projb, 256, 64, 256, 256, stream);
  transpose64x256<<<64, 256, 0, stream>>>(projb, projT);
  launch_gemm(0, Xn, 256, projb, nullptr, nullptr, nullptr, 0, scores, 64, BN, 64, 256, stream);
  softmax64<<<BN / 8, 256, 0, stream>>>(scores, BN);
  launch_gemm(0, scores, 64, projT, nullptr, nullptr, nullptr, 0, fus, 256, BN, 256, 64, stream); // ctx
  launch_gemm(2, Xn, 256, gate_w, gate_b, nullptr, nullptr, 0, wbuf, 256, BN, 256, 256, stream);  // gate
  pcm_combine<<<EB, 256, 0, stream>>>(xbuf, Xn, wbuf, fus, ls2, x2, EL);

  // ================= FFN ==================================================
  ln_kernel<<<BN / 8, 256, 0, stream>>>(x2, ln3_g, ln3_b, Xn, BN);
  launch_gemm(1, Xn, 256, ffn_w1, ffn_b1, nullptr, nullptr, 0, hbuf, 1024, BN, 1024, 256, stream);
  launch_gemm(0, hbuf, 1024, ffn_w2, ffn_b2, ls3, x2, 256, (float*)d_out, 256, BN, 256, 1024, stream);

  (void)in_sizes; (void)n_in; (void)out_size; (void)ws_size;
}
